// CNN1D_29472065585782
// MI455X (gfx1250) — matrix operand order repaired and checked by reading, not compiled, not run on hardware
//
#include <hip/hip_runtime.h>
#include <hip/hip_bf16.h>

// ---------------------------------------------------------------------------
// Conv1D (B=32, L=8192, Cin=64, K=3, F=128, VALID) + bias + ReLU for gfx1250.
// Implicit GEMM: M = B*Lout, N = 128, K = 192 (im2col rows are contiguous in x
// because channel stride is 1 and row stride is 64 == Cin).
// bf16 WMMA 16x16x32 with f32 accumulation; A staged via VGPR convert,
// B staged via GLOBAL_LOAD_ASYNC_TO_LDS (ASYNCcnt); NT stores for output.
// ---------------------------------------------------------------------------

typedef __attribute__((ext_vector_type(16))) __bf16 v16bf;
typedef __attribute__((ext_vector_type(8)))  __bf16 v8bf;
typedef __attribute__((ext_vector_type(4)))  __bf16 v4bf;
typedef __attribute__((ext_vector_type(8)))  float  v8f;

#define BATCH 32
#define LLEN  8192
#define CIN   64
#define KW    3
#define FOUT  128
#define LOUT  (LLEN - KW + 1)          // 8190
#define KDIM  (KW * CIN)               // 192

#define MT    64                        // M tile (output positions per block)
#define TILES_PER_B 128                 // ceil(8190/64)
#define LDA   72                        // padded LDS row stride for A (bf16)
#define LDB   200                       // padded LDS row stride for Bt (bf16)

// Async global->LDS 16-byte copy (CDNA5). LDS byte address = low 32 bits of
// the generic pointer (ISA 10.2 aperture rule).
__device__ __forceinline__ void async_copy_b128(void* lds_dst, const void* gsrc) {
    uint32_t l = (uint32_t)(uintptr_t)lds_dst;
    uint64_t g = (uint64_t)(uintptr_t)gsrc;
    asm volatile("global_load_async_to_lds_b128 %0, %1, off"
                 :: "v"(l), "v"(g) : "memory");
}
__device__ __forceinline__ void wait_async0() {
    asm volatile("s_wait_asynccnt 0x0" ::: "memory");
}

// ---------------------------------------------------------------------------
// Prepass: w [K=3, Cin=64, F=128] f32  ->  wt [F=128, K*Cin=192] bf16 (in ws).
// ---------------------------------------------------------------------------
__global__ void conv1d_wcvt(const float* __restrict__ w, __bf16* __restrict__ wt) {
    int i = blockIdx.x * 256 + threadIdx.x;
    if (i < FOUT * KDIM) {
        int f  = i / KDIM;
        int kk = i % KDIM;
        wt[i] = (__bf16)w[kk * FOUT + f];     // w flat: [kk][f]
    }
}

// ---------------------------------------------------------------------------
// Main kernel: one block = 64 output positions x all 128 filters.
// 8 waves: (wm:0..3, wn:0..1); each wave -> 16 rows x 64 cols
// (four 16x16 WMMA accumulators), K loop = 6 steps of 32.
// ---------------------------------------------------------------------------
__global__ __launch_bounds__(256)
void conv1d_wmma(const float* __restrict__ x,
                 const __bf16* __restrict__ wt,
                 const float* __restrict__ bias,
                 float* __restrict__ out) {
    __shared__ __bf16 shA[66 * LDA];      //  9.5 KB: x tile (66 rows x 64 ch, padded)
    __shared__ __bf16 shB[FOUT * LDB];    // 51.2 KB: wt tile (128 x 192, padded)

    const int tid  = threadIdx.x;
    const int b    = blockIdx.x >> 7;         // / TILES_PER_B
    const int tile = blockIdx.x & (TILES_PER_B - 1);
    const int l0   = tile * MT;

    // ---- Stage Bt first, asynchronously (no conversion needed) --------------
    // 3072 16-byte chunks; 12 async issues per thread, ASYNCcnt-tracked.
    for (int c = tid; c < (FOUT * KDIM) / 8; c += 256) {
        int n   = c / 24;                   // 24 chunks of 8 bf16 per row
        int off = (c % 24) * 8;
        async_copy_b128(&shB[n * LDB + off], wt + n * KDIM + off);
    }

    // ---- Stage A: 66 rows x 64 f32 -> bf16 LDS (row pad 8, overlap-safe) ----
    // Runs while the async B copies are in flight.
    const float* xb = x + ((size_t)b * LLEN + l0) * CIN;
    for (int idx = tid; idx < (66 * 64) / 4; idx += 256) {  // 1056 float4 loads
        int row = idx >> 4;                 // 16 float4 per 64-ch row
        int c4  = (idx & 15) << 2;
        float4 v = make_float4(0.f, 0.f, 0.f, 0.f);
        if (l0 + row < LLEN)
            v = *(const float4*)(xb + row * CIN + c4);
        v4bf t4 = { (__bf16)v.x, (__bf16)v.y, (__bf16)v.z, (__bf16)v.w };
        *(v4bf*)&shA[row * LDA + c4] = t4;
    }

    wait_async0();          // our async B chunks are in LDS
    __syncthreads();        // everyone's chunks visible to everyone

    // ---- Compute ------------------------------------------------------------
    const int lane = tid & 31;
    const int wave = tid >> 5;
    const int wm   = wave >> 1;             // 0..3  -> M subtile
    const int wn   = wave & 1;              // 0..1  -> N half
    const int hh   = lane >> 4;             // half-wave (K split)
    const int r    = lane & 15;             // row/col within subtile

    const int aRow = wm * 16 + r;

    v8f acc0 = {}, acc1 = {}, acc2 = {}, acc3 = {};

    #pragma unroll
    for (int kk = 0; kk < KDIM; kk += 32) {
        // A fragment: rows M=r, K chunks {kk+16h .. +7} and {kk+16h+8 .. +7}
        union { v16bf v; v8bf h[2]; } a;
        int ko0 = kk + hh * 16;
        int ko1 = ko0 + 8;
        a.h[0] = *(const v8bf*)&shA[aRow * LDA + ko0 + ((ko0 >> 6) << 3)];
        a.h[1] = *(const v8bf*)&shA[aRow * LDA + ko1 + ((ko1 >> 6) << 3)];

        // B fragments: column n, contiguous 16 bf16 at K = kk + 16h
        const int kb = kk + hh * 16;
        v16bf b0 = *(const v16bf*)&shB[(wn * 64 +  0 + r) * LDB + kb];
        v16bf b1 = *(const v16bf*)&shB[(wn * 64 + 16 + r) * LDB + kb];
        v16bf b2 = *(const v16bf*)&shB[(wn * 64 + 32 + r) * LDB + kb];
        v16bf b3 = *(const v16bf*)&shB[(wn * 64 + 48 + r) * LDB + kb];

        acc0 = __builtin_amdgcn_wmma_f32_16x16x32_bf16(false, a.v, false, b0, (short)0, acc0, false, false);
        acc1 = __builtin_amdgcn_wmma_f32_16x16x32_bf16(false, a.v, false, b1, (short)0, acc1, false, false);
        acc2 = __builtin_amdgcn_wmma_f32_16x16x32_bf16(false, a.v, false, b2, (short)0, acc2, false, false);
        acc3 = __builtin_amdgcn_wmma_f32_16x16x32_bf16(false, a.v, false, b3, (short)0, acc3, false, false);
    }

    // ---- Bias + ReLU + non-temporal store ----------------------------------
    // C/D layout: VGPR i -> M = 8*hh + i (within subtile), N = r.
    // NT stores: output is write-once; keep L2 for wt + x halos.
    const size_t outBase = ((size_t)b * LOUT + l0) * FOUT;
    v8f accs[4] = { acc0, acc1, acc2, acc3 };
    #pragma unroll
    for (int j = 0; j < 4; ++j) {
        int n   = wn * 64 + j * 16 + r;
        float bv = bias[n];
        #pragma unroll
        for (int i = 0; i < 8; ++i) {
            int m = wm * 16 + hh * 8 + i;
            if (l0 + m < LOUT) {
                float v = accs[j][i] + bv;
                __builtin_nontemporal_store(v > 0.f ? v : 0.f,
                                            &out[outBase + (size_t)m * FOUT + n]);
            }
        }
    }
}

// ---------------------------------------------------------------------------
extern "C" void kernel_launch(void* const* d_in, const int* in_sizes, int n_in,
                              void* d_out, int out_size, void* d_ws, size_t ws_size,
                              hipStream_t stream) {
    const float* x    = (const float*)d_in[0];
    const float* w    = (const float*)d_in[1];
    const float* bias = (const float*)d_in[2];
    float*       out  = (float*)d_out;
    __bf16*      wt   = (__bf16*)d_ws;     // 128*192*2 = 48 KB of scratch

    conv1d_wcvt<<<(FOUT * KDIM + 255) / 256, 256, 0, stream>>>(w, wt);
    conv1d_wmma<<<BATCH * TILES_PER_B, 256, 0, stream>>>(x, wt, bias, out);
}
